// MySumNetwork_11922829214418
// MI455X (gfx1250) — compile-verified, matrix operand order checked by reading, not run on hardware
//
#include <hip/hip_runtime.h>
#include <hip/hip_bf16.h>
#include <math.h>

typedef __attribute__((ext_vector_type(16))) _Float16 v16h;
typedef __attribute__((ext_vector_type(8)))  _Float16 v8h;
typedef __attribute__((ext_vector_type(8)))  float    v8f;

#define BATCH 4
#define NATOM 384
#define NRBF  32
#define KDIM  (NATOM * NRBF)      // 12288
#define MIDC  64
#define MLPH  128
#define NSPLIT 4                  // split-K factor for the WMMA GEMMs
#define KSL   (KDIM / NSPLIT)     // 3072 halfs per K-slice (96 WMMA steps, even)

// ---------------------------------------------------------------------------
// 0) embedding + first "in" linear (EMB==1, so it's scalar * Win1 + bin1)
__global__ void k_embed_h1pre(const int* __restrict__ feat,
                              const float* __restrict__ emb,
                              const float* __restrict__ Win1,
                              const float* __restrict__ bin1,
                              float* __restrict__ h1pre) {
    int idx = blockIdx.x * blockDim.x + threadIdx.x;   // B*N*64
    if (idx >= BATCH * NATOM * MIDC) return;
    int c = idx % MIDC;
    int j = (idx / MIDC) % NATOM;
    int b = idx / (MIDC * NATOM);
    float x = emb[feat[b * NATOM + j]];                 // EMB = 1
    h1pre[idx] = x * Win1[c] + bin1[c];
}

// ---------------------------------------------------------------------------
// 1) pairwise distances -> RBF, stored fp16 as A-matrix [b][i][k = j*32 + r]
__global__ void k_rbf(const float* __restrict__ geo, _Float16* __restrict__ rbf) {
    int idx = blockIdx.x * blockDim.x + threadIdx.x;   // B*N*N
    if (idx >= BATCH * NATOM * NATOM) return;
    int j = idx % NATOM;
    int i = (idx / NATOM) % NATOM;
    int b = idx / (NATOM * NATOM);
    const float* gi = geo + ((size_t)b * NATOM + i) * 3;
    const float* gj = geo + ((size_t)b * NATOM + j) * 3;
    float dx = gi[0] - gj[0], dy = gi[1] - gj[1], dz = gi[2] - gj[2];
    float d = sqrtf(dx * dx + dy * dy + dz * dz + 1e-12f);
    union { v8h v[4]; _Float16 h[32]; } u;
#pragma unroll
    for (int r = 0; r < NRBF; ++r) {
        float mu = (5.0f / 31.0f) * (float)r;
        float t  = d - mu;
        u.h[r] = (_Float16)__expf(-10.0f * t * t);
    }
    v8h* out = (v8h*)(rbf + ((size_t)b * NATOM + i) * KDIM + (size_t)j * NRBF);
    out[0] = u.v[0]; out[1] = u.v[1]; out[2] = u.v[2]; out[3] = u.v[3];
}

// ---------------------------------------------------------------------------
// 2) Gt[b][c][k=j*32+r] = Wr[r][c] * hpre[b][j][c]   (fp16, B-operand transposed)
__global__ void k_build_gt(const float* __restrict__ Wr,    // [32][C]
                           const float* __restrict__ hpre,  // [B][N][C]
                           _Float16* __restrict__ Gt,       // [B][C][K]
                           int C) {
    int idx = blockIdx.x * blockDim.x + threadIdx.x;   // B*C*N
    if (idx >= BATCH * C * NATOM) return;
    int j = idx % NATOM;
    int c = (idx / NATOM) % C;
    int b = idx / (NATOM * C);
    float h = hpre[((size_t)b * NATOM + j) * C + c];
    union { v8h v[4]; _Float16 h[32]; } u;
#pragma unroll
    for (int r = 0; r < NRBF; ++r)
        u.h[r] = (_Float16)(Wr[r * C + c] * h);
    v8h* out = (v8h*)(Gt + ((size_t)(b * C + c)) * KDIM + (size_t)j * NRBF);
    out[0] = u.v[0]; out[1] = u.v[1]; out[2] = u.v[2]; out[3] = u.v[3];
}

// ---------------------------------------------------------------------------
// zero-fill (output accumulators for split-K atomics)
__global__ void k_zero(float* __restrict__ p, int n) {
    int i = blockIdx.x * blockDim.x + threadIdx.x;
    if (i < n) p[i] = 0.f;
}

// ---------------------------------------------------------------------------
// fragment loader: A row chunk + 4 B column-tile chunks for one K-step
__device__ __forceinline__ void load_frags(const _Float16* __restrict__ pA,
                                           const _Float16* __restrict__ pB,
                                           int k0,
                                           v16h& a, v16h& b0, v16h& b1,
                                           v16h& b2, v16h& b3) {
    const size_t tstep = (size_t)16 * KDIM;   // next 16-column tile of Bt
    union F { v16h v; v8h h[2]; } t;
    t.h[0] = *(const v8h*)(pA + k0);
    t.h[1] = *(const v8h*)(pA + k0 + 16);
    a = t.v;
    t.h[0] = *(const v8h*)(pB + k0);
    t.h[1] = *(const v8h*)(pB + k0 + 16);
    b0 = t.v;
    t.h[0] = *(const v8h*)(pB + tstep + k0);
    t.h[1] = *(const v8h*)(pB + tstep + k0 + 16);
    b1 = t.v;
    t.h[0] = *(const v8h*)(pB + 2 * tstep + k0);
    t.h[1] = *(const v8h*)(pB + 2 * tstep + k0 + 16);
    b2 = t.v;
    t.h[0] = *(const v8h*)(pB + 3 * tstep + k0);
    t.h[1] = *(const v8h*)(pB + 3 * tstep + k0 + 16);
    b3 = t.v;
}

// ---------------------------------------------------------------------------
// 3) WMMA GEMM, 1x4 register blocking + split-K + 2-stage software pipeline.
//    __launch_bounds__(128, 1): few waves total -> trade occupancy for a big
//    register file so both fragment sets + 4 accumulator chains stay in VGPRs
//    (no scratch spills in the hot loop).
__global__ void __launch_bounds__(128, 1)
k_gemm_wmma(const _Float16* __restrict__ A,   // [B][M][K]
            const _Float16* __restrict__ Bt,  // [B][NC][K]
            float* __restrict__ Cout,         // [B][M][NC] (pre-zeroed)
            int M, int NC, int total_units, float scale) {
    int lane = threadIdx.x;                       // 0..31
    int unit = blockIdx.x * blockDim.y + threadIdx.y;
    if (unit >= total_units) return;              // wave-uniform exit
    int nblocks = NC / 64;                        // 64-column blocks
    int mtiles  = M / 16;
    int units_per_b = mtiles * nblocks * NSPLIT;
    int b  = unit / units_per_b;
    int u  = unit % units_per_b;
    int ks = u % NSPLIT;
    int u2 = u / NSPLIT;
    int nb = u2 % nblocks;
    int ti = u2 / nblocks;
    int kbeg = ks * KSL;

    // A fragment addressing: lane<16 -> row M=lane, K {0..7,16..23};
    //                        lane>=16 -> same row set, K {8..15,24..31}
    int row = lane & 15;
    int kb  = (lane >> 4) * 8;
    const _Float16* pA = A  + (size_t)b * M  * KDIM + (size_t)(ti * 16 + row) * KDIM + kbeg + kb;
    const _Float16* pB = Bt + (size_t)b * NC * KDIM + (size_t)(nb * 64 + row) * KDIM + kbeg + kb;

    v8f acc0 = {}, acc1 = {}, acc2 = {}, acc3 = {};

    // two explicit fragment sets -> software pipeline
    v16h aX, b0X, b1X, b2X, b3X;      // set X
    v16h aY, b0Y, b1Y, b2Y, b3Y;      // set Y

    load_frags(pA, pB, 0, aX, b0X, b1X, b2X, b3X);

    for (int k0 = 0; k0 < KSL - 64; k0 += 64) {
        // stage 1: issue loads for set Y (k0+32), compute on set X
        load_frags(pA, pB, k0 + 32, aY, b0Y, b1Y, b2Y, b3Y);
        acc0 = __builtin_amdgcn_wmma_f32_16x16x32_f16(false, aX, false, b0X, (short)0, acc0, false, false);
        acc1 = __builtin_amdgcn_wmma_f32_16x16x32_f16(false, aX, false, b1X, (short)0, acc1, false, false);
        acc2 = __builtin_amdgcn_wmma_f32_16x16x32_f16(false, aX, false, b2X, (short)0, acc2, false, false);
        acc3 = __builtin_amdgcn_wmma_f32_16x16x32_f16(false, aX, false, b3X, (short)0, acc3, false, false);
        // stage 2: issue loads for set X (k0+64), compute on set Y
        load_frags(pA, pB, k0 + 64, aX, b0X, b1X, b2X, b3X);
        acc0 = __builtin_amdgcn_wmma_f32_16x16x32_f16(false, aY, false, b0Y, (short)0, acc0, false, false);
        acc1 = __builtin_amdgcn_wmma_f32_16x16x32_f16(false, aY, false, b1Y, (short)0, acc1, false, false);
        acc2 = __builtin_amdgcn_wmma_f32_16x16x32_f16(false, aY, false, b2Y, (short)0, acc2, false, false);
        acc3 = __builtin_amdgcn_wmma_f32_16x16x32_f16(false, aY, false, b3Y, (short)0, acc3, false, false);
    }
    // tail: two remaining K-steps (KSL/32 is even), no OOB loads
    load_frags(pA, pB, KSL - 32, aY, b0Y, b1Y, b2Y, b3Y);
    acc0 = __builtin_amdgcn_wmma_f32_16x16x32_f16(false, aX, false, b0X, (short)0, acc0, false, false);
    acc1 = __builtin_amdgcn_wmma_f32_16x16x32_f16(false, aX, false, b1X, (short)0, acc1, false, false);
    acc2 = __builtin_amdgcn_wmma_f32_16x16x32_f16(false, aX, false, b2X, (short)0, acc2, false, false);
    acc3 = __builtin_amdgcn_wmma_f32_16x16x32_f16(false, aX, false, b3X, (short)0, acc3, false, false);
    acc0 = __builtin_amdgcn_wmma_f32_16x16x32_f16(false, aY, false, b0Y, (short)0, acc0, false, false);
    acc1 = __builtin_amdgcn_wmma_f32_16x16x32_f16(false, aY, false, b1Y, (short)0, acc1, false, false);
    acc2 = __builtin_amdgcn_wmma_f32_16x16x32_f16(false, aY, false, b2Y, (short)0, acc2, false, false);
    acc3 = __builtin_amdgcn_wmma_f32_16x16x32_f16(false, aY, false, b3Y, (short)0, acc3, false, false);

    // C/D layout: lane -> column (lane&15), VGPR v -> row v + 8*(lane>>4)
    float* Cb = Cout + (size_t)b * M * NC + (size_t)(ti * 16) * NC + nb * 64;
    int col    = lane & 15;
    int rowoff = (lane >> 4) * 8;
    v8f accs[4] = { acc0, acc1, acc2, acc3 };
#pragma unroll
    for (int t = 0; t < 4; ++t)
#pragma unroll
        for (int v = 0; v < 8; ++v)
            atomicAdd(&Cb[(size_t)(rowoff + v) * NC + t * 16 + col], accs[t][v] * scale);
}

// ---------------------------------------------------------------------------
// 4) Y = silu(X @ W + bias)     X:[rows][Cin]  W:[Cin][Cout]
__global__ void k_linear_silu(const float* __restrict__ X, const float* __restrict__ W,
                              const float* __restrict__ bias, float* __restrict__ Y,
                              int Cin, int Cout, int rows) {
    int idx = blockIdx.x * blockDim.x + threadIdx.x;
    if (idx >= rows * Cout) return;
    int c = idx % Cout;
    int r = idx / Cout;
    const float* x = X + (size_t)r * Cin;
    float s = bias[c];
    for (int k = 0; k < Cin; ++k) s += x[k] * W[(size_t)k * Cout + c];
    Y[idx] = s / (1.0f + __expf(-s));
}

// 5) Y = X @ W + bias
__global__ void k_linear(const float* __restrict__ X, const float* __restrict__ W,
                         const float* __restrict__ bias, float* __restrict__ Y,
                         int Cin, int Cout, int rows) {
    int idx = blockIdx.x * blockDim.x + threadIdx.x;
    if (idx >= rows * Cout) return;
    int c = idx % Cout;
    int r = idx / Cout;
    const float* x = X + (size_t)r * Cin;
    float s = bias[c];
    for (int k = 0; k < Cin; ++k) s += x[k] * W[(size_t)k * Cout + c];
    Y[idx] = s;
}

// ---------------------------------------------------------------------------
// 6) BatchNorm over (batch, feature) per atom, then leaky-relu (in-place)
__global__ void k_bn_lrelu(float* __restrict__ X, const float* __restrict__ g,
                           const float* __restrict__ beta, int F) {
    int n   = blockIdx.x;           // atom index 0..383
    int tid = threadIdx.x;          // 128 threads
    __shared__ float ssum[128], ssq[128];
    int cnt = BATCH * F;
    float s = 0.f, q = 0.f;
    for (int e = tid; e < cnt; e += blockDim.x) {
        int b = e / F, f = e % F;
        float v = X[((size_t)b * NATOM + n) * F + f];
        s += v; q += v * v;
    }
    ssum[tid] = s; ssq[tid] = q;
    __syncthreads();
    for (int st = 64; st > 0; st >>= 1) {
        if (tid < st) { ssum[tid] += ssum[tid + st]; ssq[tid] += ssq[tid + st]; }
        __syncthreads();
    }
    float mu  = ssum[0] / (float)cnt;
    float var = ssq[0] / (float)cnt - mu * mu;
    float inv = rsqrtf(var + 1e-5f);
    float gg = g[n], bb = beta[n];
    for (int e = tid; e < cnt; e += blockDim.x) {
        int b = e / F, f = e % F;
        size_t off = ((size_t)b * NATOM + n) * F + f;
        float v = (X[off] - mu) * inv * gg + bb;
        X[off] = v >= 0.f ? v : 0.2f * v;
    }
}

// ---------------------------------------------------------------------------
// 7) out[b][f] = sqrt(sum_n Y[b][n][f]^2)
__global__ void k_lp_pool(const float* __restrict__ Y, float* __restrict__ out, int F) {
    int idx = blockIdx.x * blockDim.x + threadIdx.x;   // B*F
    if (idx >= BATCH * F) return;
    int f = idx % F, b = idx / F;
    float s = 0.f;
    for (int n = 0; n < NATOM; ++n) {
        float v = Y[((size_t)b * NATOM + n) * F + f];
        s += v * v;
    }
    out[idx] = sqrtf(s);
}

// ===========================================================================
extern "C" void kernel_launch(void* const* d_in, const int* in_sizes, int n_in,
                              void* d_out, int out_size, void* d_ws, size_t ws_size,
                              hipStream_t stream) {
    const int*   feat  = (const int*)  d_in[0];
    const float* geo   = (const float*)d_in[1];
    const float* emb   = (const float*)d_in[2];
    const float* Wr1   = (const float*)d_in[3];
    const float* Win1  = (const float*)d_in[4];
    const float* bin1  = (const float*)d_in[5];
    const float* Wout1 = (const float*)d_in[6];
    const float* bout1 = (const float*)d_in[7];
    const float* Wr2   = (const float*)d_in[8];
    const float* Win2  = (const float*)d_in[9];
    const float* bin2  = (const float*)d_in[10];
    const float* Wout2 = (const float*)d_in[11];
    const float* bout2 = (const float*)d_in[12];
    const float* W1    = (const float*)d_in[13];
    const float* b1    = (const float*)d_in[14];
    const float* g1    = (const float*)d_in[15];
    const float* beta1 = (const float*)d_in[16];
    const float* W2    = (const float*)d_in[17];
    const float* b2    = (const float*)d_in[18];
    const float* g2    = (const float*)d_in[19];
    const float* beta2 = (const float*)d_in[20];
    float* out = (float*)d_out;

    // workspace carve-up (bytes)
    char* ws = (char*)d_ws;
    size_t off = 0;
    auto alloc = [&](size_t bytes) { char* p = ws + off; off = (off + bytes + 255) & ~(size_t)255; return p; };
    _Float16* rbf   = (_Float16*)alloc((size_t)BATCH * NATOM * KDIM * sizeof(_Float16)); // 37.75 MB
    _Float16* Gt    = (_Float16*)alloc((size_t)BATCH * MLPH  * KDIM * sizeof(_Float16)); // 12.58 MB (shared by both layers)
    float* h1pre    = (float*)alloc((size_t)BATCH * NATOM * MIDC * sizeof(float));
    float* h1       = (float*)alloc((size_t)BATCH * NATOM * MIDC * sizeof(float));
    float* h2pre    = (float*)alloc((size_t)BATCH * NATOM * MLPH * sizeof(float));
    float* mbuf     = (float*)alloc((size_t)BATCH * NATOM * MLPH * sizeof(float));
    float* h2       = (float*)alloc((size_t)BATCH * NATOM * MLPH * sizeof(float));
    float* y1       = (float*)alloc((size_t)BATCH * NATOM * MLPH * sizeof(float));
    float* y2       = (float*)alloc((size_t)BATCH * NATOM * 2 * MLPH * sizeof(float));

    const float scale = 1.0f / sqrtf((float)NATOM);
    const int rows = BATCH * NATOM;

    // 0) embed + in-linear 1
    {
        int tot = BATCH * NATOM * MIDC;
        k_embed_h1pre<<<(tot + 255) / 256, 256, 0, stream>>>(feat, emb, Win1, bin1, h1pre);
    }
    // 1) rbf
    {
        int tot = BATCH * NATOM * NATOM;
        k_rbf<<<(tot + 255) / 256, 256, 0, stream>>>(geo, rbf);
    }
    // 2) Gt1
    {
        int tot = BATCH * MIDC * NATOM;
        k_build_gt<<<(tot + 255) / 256, 256, 0, stream>>>(Wr1, h1pre, Gt, MIDC);
    }
    // 3) GEMM1: [384 x 12288] x [12288 x 64] per batch, split-K=4 + fp32 atomics
    {
        int tot = BATCH * NATOM * MIDC;
        k_zero<<<(tot + 255) / 256, 256, 0, stream>>>(mbuf, tot);
        int total_units = BATCH * (NATOM / 16) * (MIDC / 64) * NSPLIT;  // 384
        dim3 blk(32, 4);
        k_gemm_wmma<<<(total_units + 3) / 4, blk, 0, stream>>>(rbf, Gt, mbuf,
                                                               NATOM, MIDC, total_units, scale);
    }
    // 4) h1 = silu(m1 @ Wout1 + bout1)
    {
        int tot = rows * MIDC;
        k_linear_silu<<<(tot + 255) / 256, 256, 0, stream>>>(mbuf, Wout1, bout1, h1, MIDC, MIDC, rows);
    }
    // 5) h2pre = h1 @ Win2 + bin2
    {
        int tot = rows * MLPH;
        k_linear<<<(tot + 255) / 256, 256, 0, stream>>>(h1, Win2, bin2, h2pre, MIDC, MLPH, rows);
    }
    // 6) Gt2
    {
        int tot = BATCH * MLPH * NATOM;
        k_build_gt<<<(tot + 255) / 256, 256, 0, stream>>>(Wr2, h2pre, Gt, MLPH);
    }
    // 7) GEMM2: [384 x 12288] x [12288 x 128] per batch, split-K=4 + fp32 atomics
    {
        int tot = BATCH * NATOM * MLPH;
        k_zero<<<(tot + 255) / 256, 256, 0, stream>>>(mbuf, tot);
        int total_units = BATCH * (NATOM / 16) * (MLPH / 64) * NSPLIT;  // 768
        dim3 blk(32, 4);
        k_gemm_wmma<<<(total_units + 3) / 4, blk, 0, stream>>>(rbf, Gt, mbuf,
                                                               NATOM, MLPH, total_units, scale);
    }
    // 8) h2 = silu(m2 @ Wout2 + bout2)
    {
        int tot = rows * MLPH;
        k_linear_silu<<<(tot + 255) / 256, 256, 0, stream>>>(mbuf, Wout2, bout2, h2, MLPH, MLPH, rows);
    }
    // 9) y1 = h2 @ W1 + b1
    {
        int tot = rows * MLPH;
        k_linear<<<(tot + 255) / 256, 256, 0, stream>>>(h2, W1, b1, y1, MLPH, MLPH, rows);
    }
    // 10) BN(atoms) + leaky (in place)
    k_bn_lrelu<<<NATOM, 128, 0, stream>>>(y1, g1, beta1, MLPH);
    // 11) y2 = y1 @ W2 + b2
    {
        int tot = rows * 2 * MLPH;
        k_linear<<<(tot + 255) / 256, 256, 0, stream>>>(y1, W2, b2, y2, MLPH, 2 * MLPH, rows);
    }
    // 12) BN(atoms) + leaky (in place)
    k_bn_lrelu<<<NATOM, 128, 0, stream>>>(y2, g2, beta2, 2 * MLPH);
    // 13) L2 pool over atoms -> [4, 256]
    {
        int tot = BATCH * 2 * MLPH;
        k_lp_pool<<<(tot + 255) / 256, 256, 0, stream>>>(y2, out, 2 * MLPH);
    }
}
